// EncodeLayer_25194278158712
// MI455X (gfx1250) — matrix operand order repaired and checked by reading, not compiled, not run on hardware
//
#include <hip/hip_runtime.h>

// ---------------------------------------------------------------------------
// CDNA5 (gfx1250) transformer encoder layer, bf16 WMMA everywhere.
// ---------------------------------------------------------------------------

typedef __attribute__((ext_vector_type(16))) __bf16 v16bf;
typedef __attribute__((ext_vector_type(4)))  __bf16 v4bf;
typedef __attribute__((ext_vector_type(8)))  float  v8f;

#define S_LEN 2048
#define D_MODEL 1024
#define N_HEADS 16
#define D_HEAD 64

// ------------------------------- fp32 -> bf16 ------------------------------
__global__ void cvt_bf16_kernel(const float4* __restrict__ in,
                                v4bf* __restrict__ out, int n4) {
  int i = blockIdx.x * blockDim.x + threadIdx.x;
  if (i < n4) {
    float4 f = in[i];
    v4bf o;
    o[0] = (__bf16)f.x; o[1] = (__bf16)f.y;
    o[2] = (__bf16)f.z; o[3] = (__bf16)f.w;
    out[i] = o;
  }
}

// ------------------------------- GEMM (WMMA) -------------------------------
// C[M,N] = relu?( alpha * A[M,K] @ B[K,N] + bias ), A/B bf16 row-major,
// fp32 accumulate. Epilogue configuration is compile-time (branch-free).
#define BM 128
#define BN 64
#define BK 32
#define LDA_S 40   // LDS stride (bf16 elems), 80B keeps 16B alignment
#define LDB_S 40

template <bool HAS_BIAS, bool RELU, bool OUTF, bool OUTB>
__global__ __launch_bounds__(256)
void gemm_bf16_kernel(const __bf16* __restrict__ A, const __bf16* __restrict__ B,
                      float* __restrict__ Cf, __bf16* __restrict__ Cb,
                      const float* __restrict__ bias,
                      int M, int N, int K, float alpha) {
  __shared__ __bf16 As[BM * LDA_S];
  __shared__ __bf16 Bs[BN * LDB_S];

  const int tid  = threadIdx.x;
  const int lane = tid & 31;
  const int wid  = tid >> 5;
  const int wm   = wid >> 1;          // 0..3 -> 32-row strip
  const int wn   = wid & 1;           // 0..1 -> 32-col strip
  const int hl   = lane >> 4;         // half-wave select
  const int l15  = lane & 15;
  const int bm   = blockIdx.y * BM;
  const int bn   = blockIdx.x * BN;

  // loaders
  const int ar  = tid >> 1;           // A row 0..127
  const int ac  = (tid & 1) * 16;     // A col chunk (16 bf16 = 32B)
  const int bk  = tid >> 3;           // B k-row 0..31
  const int bn8 = (tid & 7) * 8;      // B col chunk of 8

  v8f acc[2][2] = {};

  // register-staged tile (double buffer: global->reg overlaps WMMA)
  uint4 ra0, ra1, rb0;
  {
    const __bf16* sa = A + (size_t)(bm + ar) * K + ac;
    ra0 = *(const uint4*)(sa);
    ra1 = *(const uint4*)(sa + 8);
    const __bf16* sb = B + (size_t)bk * N + bn + bn8;
    rb0 = *(const uint4*)(sb);
  }

  for (int k0 = 0; k0 < K; k0 += BK) {
    // commit staged registers to LDS
    *(uint4*)(&As[ar * LDA_S + ac])     = ra0;
    *(uint4*)(&As[ar * LDA_S + ac + 8]) = ra1;
    {
      __bf16 t[8];
      *(uint4*)t = rb0;
#pragma unroll
      for (int j = 0; j < 8; ++j) Bs[(bn8 + j) * LDB_S + bk] = t[j];
    }
    __syncthreads();

    // issue next tile's global loads (in flight during WMMAs below)
    if (k0 + BK < K) {
      const __bf16* sa = A + (size_t)(bm + ar) * K + (k0 + BK) + ac;
      ra0 = *(const uint4*)(sa);
      ra1 = *(const uint4*)(sa + 8);
      const __bf16* sb = B + (size_t)(k0 + BK + bk) * N + bn + bn8;
      rb0 = *(const uint4*)(sb);
    }
    if (k0 + 2 * BK < K) {  // pull tile k0+2*BK toward GL2
      __builtin_prefetch(A + (size_t)(bm + ar) * K + (k0 + 2 * BK) + ac, 0, 1);
      __builtin_prefetch(B + (size_t)(k0 + 2 * BK + bk) * N + bn + bn8, 0, 1);
    }

    v16bf af[2], bf[2];
#pragma unroll
    for (int t = 0; t < 2; ++t) {
      // A-frag: lane holds row M=l15; K = {kb..kb+7, kb+16..kb+23}, kb=8*hl
      const __bf16* pa = &As[(wm * 32 + t * 16 + l15) * LDA_S + hl * 8];
      union { v16bf v; uint4 q[2]; } ua;
      ua.q[0] = *(const uint4*)(pa);
      ua.q[1] = *(const uint4*)(pa + 16);
      af[t] = ua.v;
      // B-frag: lane holds col N=l15; K = {kb..kb+7, kb+16..kb+23}, kb=8*hl
      const __bf16* pb = &Bs[(wn * 32 + t * 16 + l15) * LDB_S + hl * 8];
      union { v16bf v; uint4 q[2]; } ub;
      ub.q[0] = *(const uint4*)(pb);
      ub.q[1] = *(const uint4*)(pb + 16);
      bf[t] = ub.v;
    }
#pragma unroll
    for (int tm = 0; tm < 2; ++tm)
#pragma unroll
      for (int tn = 0; tn < 2; ++tn)
        acc[tm][tn] = __builtin_amdgcn_wmma_f32_16x16x32_bf16(
            false, af[tm], false, bf[tn], (short)0, acc[tm][tn], false, false);
    __syncthreads();
  }

  // epilogue: C layout — VGPR r -> row r + 8*hl, col = l15 (branch-free)
#pragma unroll
  for (int tm = 0; tm < 2; ++tm)
#pragma unroll
    for (int tn = 0; tn < 2; ++tn) {
      const int col = bn + wn * 32 + tn * 16 + l15;
      float bv = 0.0f;
      if constexpr (HAS_BIAS) bv = bias[col];
      const int row0 = bm + wm * 32 + tm * 16 + hl * 8;
#pragma unroll
      for (int r = 0; r < 8; ++r) {
        float v = acc[tm][tn][r] * alpha + bv;
        if constexpr (RELU) v = fmaxf(v, 0.0f);
        size_t idx = (size_t)(row0 + r) * N + col;
        if constexpr (OUTF) Cf[idx] = v;
        if constexpr (OUTB) Cb[idx] = (__bf16)v;
      }
    }
}

// --------------------------- flash attention (WMMA) ------------------------
// One workgroup: one (b,h) and 128 query rows; 8 waves x 16 rows each.
// Q is pre-scaled by 1/sqrt(d). Layouts: q/k/v/o bf16 [B, S, H*64].
#define AKV 64
#define KS 72   // LDS stride (bf16), 144B keeps 16B alignment

__global__ __launch_bounds__(256)
void attn_kernel(const __bf16* __restrict__ Q, const __bf16* __restrict__ Kg,
                 const __bf16* __restrict__ Vg, __bf16* __restrict__ O) {
  __shared__ __bf16 Kl[AKV * KS];        // [key][d]
  __shared__ __bf16 Vl[D_HEAD * KS];     // [d][key]  (transposed)
  __shared__ __bf16 Pl[8 * 16 * KS];     // per-wave P tiles [16][KS]

  const int tid  = threadIdx.x;
  const int lane = tid & 31;
  const int wid  = tid >> 5;
  const int hl   = lane >> 4;
  const int l15  = lane & 15;
  const int bh   = blockIdx.y;
  const int b    = bh >> 4;
  const int h    = bh & 15;
  const int q0   = blockIdx.x * 128 + wid * 16;
  const size_t base = ((size_t)b * S_LEN) * D_MODEL + h * D_HEAD;

  // Q A-fragments straight from global (row = q0 + l15)
  v16bf qf[2];
  {
    const __bf16* qp = Q + base + (size_t)(q0 + l15) * D_MODEL;
#pragma unroll
    for (int f = 0; f < 2; ++f) {
      int kb = f * 32 + hl * 8;
      union { v16bf v; uint4 q[2]; } u;
      u.q[0] = *(const uint4*)(qp + kb);
      u.q[1] = *(const uint4*)(qp + kb + 16);
      qf[f] = u.v;
    }
  }

  float m_i[8], l_i[8];
  v8f o[4] = {};
#pragma unroll
  for (int r = 0; r < 8; ++r) { m_i[r] = -1e30f; l_i[r] = 0.0f; }

  for (int j0 = 0; j0 < S_LEN; j0 += AKV) {
    __syncthreads();
    { // stage K (row-major) and V (transposed) tiles: 64 keys x 64 dims each
      int key = tid >> 2, dc = (tid & 3) * 16;
      const __bf16* kp = Kg + base + (size_t)(j0 + key) * D_MODEL + dc;
      uint4 a0 = *(const uint4*)(kp);
      uint4 a1 = *(const uint4*)(kp + 8);
      *(uint4*)(&Kl[key * KS + dc])     = a0;
      *(uint4*)(&Kl[key * KS + dc + 8]) = a1;
      const __bf16* vp = Vg + base + (size_t)(j0 + key) * D_MODEL + dc;
      __bf16 t[16];
      *(uint4*)(t)     = *(const uint4*)(vp);
      *(uint4*)(t + 8) = *(const uint4*)(vp + 8);
#pragma unroll
      for (int j = 0; j < 16; ++j) Vl[(dc + j) * KS + key] = t[j];
    }
    __syncthreads();

    // scores: S = Q @ K^T  (M=16 rows, N=64 keys, K=64 dims)
    v8f s[4] = {};
#pragma unroll
    for (int f = 0; f < 2; ++f)
#pragma unroll
      for (int nt = 0; nt < 4; ++nt) {
        const __bf16* pb = &Kl[(nt * 16 + l15) * KS + f * 32 + hl * 8];
        union { v16bf v; uint4 q[2]; } ub;
        ub.q[0] = *(const uint4*)(pb);
        ub.q[1] = *(const uint4*)(pb + 16);
        s[nt] = __builtin_amdgcn_wmma_f32_16x16x32_bf16(
            false, qf[f], false, ub.v, (short)0, s[nt], false, false);
      }

    // online softmax (row = r + 8*hl lives in one 16-lane half)
    __bf16* Pw = &Pl[wid * 16 * KS];
#pragma unroll
    for (int r = 0; r < 8; ++r) {
      float mx = fmaxf(fmaxf(s[0][r], s[1][r]), fmaxf(s[2][r], s[3][r]));
#pragma unroll
      for (int msk = 1; msk <= 8; msk <<= 1) mx = fmaxf(mx, __shfl_xor(mx, msk, 32));
      float mn   = fmaxf(m_i[r], mx);
      float corr = __expf(m_i[r] - mn);
      float rsum = 0.0f;
#pragma unroll
      for (int nt = 0; nt < 4; ++nt) {
        float p = __expf(s[nt][r] - mn);
        Pw[(r + hl * 8) * KS + nt * 16 + l15] = (__bf16)p;
        rsum += p;
      }
#pragma unroll
      for (int msk = 1; msk <= 8; msk <<= 1) rsum += __shfl_xor(rsum, msk, 32);
      l_i[r] = l_i[r] * corr + rsum;
      m_i[r] = mn;
#pragma unroll
      for (int vt = 0; vt < 4; ++vt) o[vt][r] *= corr;
    }
    __syncthreads();

    // O += P @ V  (A = P from LDS, B = V transposed tile)
    v16bf pf[2];
#pragma unroll
    for (int f = 0; f < 2; ++f) {
      const __bf16* pp = &Pw[l15 * KS + f * 32 + hl * 8];
      union { v16bf v; uint4 q[2]; } up;
      up.q[0] = *(const uint4*)(pp);
      up.q[1] = *(const uint4*)(pp + 16);
      pf[f] = up.v;
    }
#pragma unroll
    for (int f = 0; f < 2; ++f)
#pragma unroll
      for (int vt = 0; vt < 4; ++vt) {
        const __bf16* pv = &Vl[(vt * 16 + l15) * KS + f * 32 + hl * 8];
        union { v16bf v; uint4 q[2]; } uv;
        uv.q[0] = *(const uint4*)(pv);
        uv.q[1] = *(const uint4*)(pv + 16);
        o[vt] = __builtin_amdgcn_wmma_f32_16x16x32_bf16(
            false, pf[f], false, uv.v, (short)0, o[vt], false, false);
      }
  }

  // normalize + store context
#pragma unroll
  for (int vt = 0; vt < 4; ++vt)
#pragma unroll
    for (int r = 0; r < 8; ++r) {
      int qr = q0 + r + hl * 8;
      float val = o[vt][r] / l_i[r];
      O[base + (size_t)qr * D_MODEL + vt * 16 + l15] = (__bf16)val;
    }
}

// ------------------------- residual add + LayerNorm ------------------------
template <bool OUTB>
__global__ __launch_bounds__(256)
void ln_kernel(const float* __restrict__ x, const float* __restrict__ y,
               const float* __restrict__ g, const float* __restrict__ be,
               float* __restrict__ outf, __bf16* __restrict__ outb) {
  const int row = blockIdx.x;
  const int tid = threadIdx.x;
  const int lane = tid & 31, wid = tid >> 5;
  __shared__ float red1[8], red2[8];

  const float* xr = x + (size_t)row * D_MODEL;
  const float* yr = y + (size_t)row * D_MODEL;
  float v[4];
  float s = 0.0f;
#pragma unroll
  for (int j = 0; j < 4; ++j) {
    int i = j * 256 + tid;
    v[j] = xr[i] + yr[i];
    s += v[j];
  }
#pragma unroll
  for (int m = 16; m >= 1; m >>= 1) s += __shfl_xor(s, m, 32);
  if (lane == 0) red1[wid] = s;
  __syncthreads();
  float tot = 0.0f;
#pragma unroll
  for (int i = 0; i < 8; ++i) tot += red1[i];
  const float mean = tot * (1.0f / D_MODEL);

  float s2 = 0.0f;
#pragma unroll
  for (int j = 0; j < 4; ++j) { float d = v[j] - mean; s2 += d * d; }
#pragma unroll
  for (int m = 16; m >= 1; m >>= 1) s2 += __shfl_xor(s2, m, 32);
  if (lane == 0) red2[wid] = s2;
  __syncthreads();
  float tot2 = 0.0f;
#pragma unroll
  for (int i = 0; i < 8; ++i) tot2 += red2[i];
  const float rstd = rsqrtf(tot2 * (1.0f / D_MODEL) + 1e-5f);

#pragma unroll
  for (int j = 0; j < 4; ++j) {
    int i = j * 256 + tid;
    float ov = (v[j] - mean) * rstd * g[i] + be[i];
    outf[(size_t)row * D_MODEL + i] = ov;
    if constexpr (OUTB) outb[(size_t)row * D_MODEL + i] = (__bf16)ov;
  }
}

// --------------------------------- driver ----------------------------------
extern "C" void kernel_launch(void* const* d_in, const int* in_sizes, int n_in,
                              void* d_out, int out_size, void* d_ws, size_t ws_size,
                              hipStream_t stream) {
  const float* x    = (const float*)d_in[0];
  const float* wq   = (const float*)d_in[1];
  const float* wk   = (const float*)d_in[2];
  const float* wv   = (const float*)d_in[3];
  const float* wo   = (const float*)d_in[4];
  const float* w1   = (const float*)d_in[5];
  const float* b1   = (const float*)d_in[6];
  const float* w2   = (const float*)d_in[7];
  const float* b2   = (const float*)d_in[8];
  const float* ln1g = (const float*)d_in[9];
  const float* ln1b = (const float*)d_in[10];
  const float* ln2g = (const float*)d_in[11];
  const float* ln2b = (const float*)d_in[12];

  const int M = 2 * S_LEN;             // 4096 rows (B*S)
  const size_t MB = 1024ull * 1024ull;
  char* ws = (char*)d_ws;
  // stream-ordered buffer reuse
  __bf16* xb   = (__bf16*)(ws + 0 * MB);    // later reused as ctxb
  __bf16* wbuf = (__bf16*)(ws + 8 * MB);    // staged bf16 weight (max 8MB)
  __bf16* qb   = (__bf16*)(ws + 16 * MB);   // later reused as hb
  __bf16* kb   = (__bf16*)(ws + 24 * MB);
  __bf16* vb   = (__bf16*)(ws + 32 * MB);
  __bf16* rb   = (__bf16*)(ws + 40 * MB);   // relu(ffn1) bf16, 32MB
  float*  tmp  = (float*)(ws + 72 * MB);    // atn_out then ffn2_out, 16MB
  float*  hf   = (float*)(ws + 88 * MB);    // LN1 output fp32, 16MB
  __bf16* ctxb = xb;
  __bf16* hb   = qb;

  auto cvt = [&](const float* src, __bf16* dst, int n) {
    int n4 = n / 4;
    cvt_bf16_kernel<<<(n4 + 255) / 256, 256, 0, stream>>>(
        (const float4*)src, (v4bf*)dst, n4);
  };

  dim3 blk(256);

  // QKV projections (1/sqrt(64) folded into Q), bf16 outputs
  cvt(x, xb, M * D_MODEL);
  cvt(wq, wbuf, D_MODEL * D_MODEL);
  gemm_bf16_kernel<false, false, false, true>
      <<<dim3(D_MODEL / BN, M / BM), blk, 0, stream>>>(
          xb, wbuf, nullptr, qb, nullptr, M, D_MODEL, D_MODEL, 0.125f);
  cvt(wk, wbuf, D_MODEL * D_MODEL);
  gemm_bf16_kernel<false, false, false, true>
      <<<dim3(D_MODEL / BN, M / BM), blk, 0, stream>>>(
          xb, wbuf, nullptr, kb, nullptr, M, D_MODEL, D_MODEL, 1.0f);
  cvt(wv, wbuf, D_MODEL * D_MODEL);
  gemm_bf16_kernel<false, false, false, true>
      <<<dim3(D_MODEL / BN, M / BM), blk, 0, stream>>>(
          xb, wbuf, nullptr, vb, nullptr, M, D_MODEL, D_MODEL, 1.0f);

  // attention (x no longer needed in bf16 -> ctxb aliases xb)
  attn_kernel<<<dim3(S_LEN / 128, 2 * N_HEADS), blk, 0, stream>>>(qb, kb, vb, ctxb);

  // output projection -> tmp (fp32)
  cvt(wo, wbuf, D_MODEL * D_MODEL);
  gemm_bf16_kernel<false, false, true, false>
      <<<dim3(D_MODEL / BN, M / BM), blk, 0, stream>>>(
          ctxb, wbuf, tmp, nullptr, nullptr, M, D_MODEL, D_MODEL, 1.0f);

  // add & norm 1 -> hf (fp32) + hb (bf16)
  ln_kernel<true><<<M, blk, 0, stream>>>(x, tmp, ln1g, ln1b, hf, hb);

  // FFN1: relu(h @ w1 + b1) -> rb (bf16)
  cvt(w1, wbuf, D_MODEL * 4096);
  gemm_bf16_kernel<true, true, false, true>
      <<<dim3(4096 / BN, M / BM), blk, 0, stream>>>(
          hb, wbuf, nullptr, rb, b1, M, 4096, D_MODEL, 1.0f);
  // FFN2: rb @ w2 + b2 -> tmp (fp32)
  cvt(w2, wbuf, 4096 * D_MODEL);
  gemm_bf16_kernel<true, false, true, false>
      <<<dim3(D_MODEL / BN, M / BM), blk, 0, stream>>>(
          rb, wbuf, tmp, nullptr, b2, M, D_MODEL, 4096, 1.0f);

  // add & norm 2 -> d_out (fp32)
  ln_kernel<false><<<M, blk, 0, stream>>>(hf, tmp, ln2g, ln2b, (float*)d_out, nullptr);
}
